// PPM_30528627539980
// MI455X (gfx1250) — matrix operand order repaired and checked by reading, not compiled, not run on hardware
//
#include <hip/hip_runtime.h>

// ---------------------------------------------------------------------------
// PPM (pixel propagation) fused kernel for gfx1250 (MI455X, wave32, WMMA).
//   t  = x @ W                      (25088x256x256 GEMM, bf16 WMMA)
//   xn = l2_normalize(x, axis=C)
//   y  = relu(xn xn^T)^2 @ t        (flash-attention-style fusion)
// Round 4: fix async builtin arg types -- toolchain declares
//   __builtin_amdgcn_global_load_async_to_lds_b128(v4i*, v4i*, Ii, Ii)
// with generic (no-AS) int32x4 pointers.
// ---------------------------------------------------------------------------

constexpr int kC    = 256;
constexpr int kHW   = 3136;          // 56*56
constexpr int kB    = 8;
constexpr int kRows = kB * kHW;      // 25088

typedef __attribute__((ext_vector_type(16))) __bf16 v16bf;
typedef __attribute__((ext_vector_type(8)))  __bf16 v8bf;
typedef __attribute__((ext_vector_type(8)))  float  v8f;
typedef __attribute__((ext_vector_type(4)))  int    v4i;

__device__ __forceinline__ __bf16 f2bf(float f) { return (__bf16)f; }
__device__ __forceinline__ v8bf v8f2bf(v8f v) {
    return __builtin_convertvector(v, v8bf);
}

// ---------------- async global->LDS (ASYNCcnt path), guarded --------------
#if __has_builtin(__builtin_amdgcn_global_load_async_to_lds_b128)
#define PPM_HAS_ASYNC 1
#else
#define PPM_HAS_ASYNC 0
#endif

__device__ __forceinline__ void async_copy_b128(const __bf16* g, __bf16* l) {
#if PPM_HAS_ASYNC
    // param types per compiler diagnostic: (v4i*, v4i*, const int, const int)
    __builtin_amdgcn_global_load_async_to_lds_b128((v4i*)g, (v4i*)l, 0, 0);
#else
    *(v8bf*)l = *(const v8bf*)g;     // synchronous fallback, same structure
#endif
}

template <int N>
__device__ __forceinline__ void ppm_wait_async() {
#if PPM_HAS_ASYNC
#if __has_builtin(__builtin_amdgcn_s_wait_asynccnt)
    __builtin_amdgcn_s_wait_asynccnt(N);
#else
    asm volatile("s_wait_asynccnt %0" :: "i"(N) : "memory");
#endif
#endif
}

// A-fragment (16x32 bf16, row-major source, stride ldm elements).
// lane<16: row=lane,    K = {kk+0..7, kk+16..23}
// lane>=16: row=lane-16, K = {kk+8..15, kk+24..31}
__device__ __forceinline__ v16bf load_a_g(const __bf16* base, int ldm,
                                          int rowbase, int kk) {
    int l = threadIdx.x & 31;
    int half = l >> 4, r = l & 15;
    const __bf16* p = base + (size_t)(rowbase + r) * ldm + kk + half * 8;
    v8bf lo = *(const v8bf*)p;
    v8bf hi = *(const v8bf*)(p + 16);
    return __builtin_shufflevector(lo, hi, 0,1,2,3,4,5,6,7,
                                          8,9,10,11,12,13,14,15);
}

// B-fragment (32x16 bf16). Source holds columns contiguously:
// base[(colbase+n)*ldm + k].  lane n (n<16): col n, K = {kk+0..7, kk+16..23};
// lane n+16: col n, K = {kk+8..15, kk+24..31} (two b128 loads).
__device__ __forceinline__ v16bf load_b_g(const __bf16* base, size_t ldm,
                                          int colbase, int kk) {
    int l = threadIdx.x & 31;
    int half = l >> 4, c = l & 15;
    const __bf16* p = base + (size_t)(colbase + c) * ldm + kk + half * 8;
    v8bf lo = *(const v8bf*)p;
    v8bf hi = *(const v8bf*)(p + 16);
    return __builtin_shufflevector(lo, hi, 0,1,2,3,4,5,6,7,
                                          8,9,10,11,12,13,14,15);
}

// ---------------------------------------------------------------------------
// Kernel 1: per-pixel L2 normalize, f32 -> bf16.  One wave per pixel row.
// ---------------------------------------------------------------------------
__global__ void __launch_bounds__(256) ppm_norm_kernel(
        const float* __restrict__ x, __bf16* __restrict__ xn) {
    int wid  = threadIdx.x >> 5;
    int lane = threadIdx.x & 31;
    int row  = blockIdx.x * 8 + wid;          // 3136 blocks * 8 waves = 25088
    const float* p = x + (size_t)row * kC + lane * 8;
    v8f v = *(const v8f*)p;
    float s = 0.f;
    #pragma unroll
    for (int i = 0; i < 8; ++i) s += v[i] * v[i];
    #pragma unroll
    for (int m = 16; m; m >>= 1) s += __shfl_xor(s, m, 32);
    float inv = rsqrtf(fmaxf(s, 1e-12f));
    *(v8bf*)(xn + (size_t)row * kC + lane * 8) = v8f2bf(v * inv);
}

// ---------------------------------------------------------------------------
// Kernel 2: Wt[n][c] = bf16(W[c][n])  (so B-fragments are contiguous loads)
// ---------------------------------------------------------------------------
__global__ void ppm_wt_kernel(const float* __restrict__ W,
                              __bf16* __restrict__ Wt) {
    int n = blockIdx.x;
    int c = threadIdx.x;
    Wt[(size_t)n * kC + c] = f2bf(W[(size_t)c * kC + n]);
}

// ---------------------------------------------------------------------------
// Kernel 3: t = x @ W via bf16 WMMA; store TRANSPOSED: tT[b][c][hw]
// so the attention kernel's V blocks (K over pixels) are contiguous.
// ---------------------------------------------------------------------------
__global__ void __launch_bounds__(128) ppm_gemmt_kernel(
        const float* __restrict__ x, const __bf16* __restrict__ Wt,
        __bf16* __restrict__ tT) {
    int w = threadIdx.x >> 5;
    int l = threadIdx.x & 31;
    int half = l >> 4, r = l & 15;
    int mbase = blockIdx.x * 16;              // 16-row tiles never straddle b
    int b    = mbase / kHW;
    int mloc = mbase - b * kHW;

    v16bf afrag[8];
    #pragma unroll
    for (int kc = 0; kc < 8; ++kc) {
        int kk = kc * 32;
        const float* p = x + (size_t)(mbase + r) * kC + kk + half * 8;
        v8bf lo = v8f2bf(*(const v8f*)p);
        v8bf hi = v8f2bf(*(const v8f*)(p + 16));
        afrag[kc] = __builtin_shufflevector(lo, hi, 0,1,2,3,4,5,6,7,
                                                   8,9,10,11,12,13,14,15);
    }

    #pragma unroll
    for (int nt = 0; nt < 4; ++nt) {
        int nbase = (w * 4 + nt) * 16;
        v8f acc = {};
        #pragma unroll
        for (int kc = 0; kc < 8; ++kc) {
            v16bf bfrag = load_b_g(Wt, kC, nbase, kc * 32);
            acc = __builtin_amdgcn_wmma_f32_16x16x32_bf16(
                      false, afrag[kc], false, bfrag, (short)0, acc, false, false);
        }
        // D element (VGPR i, lane l): M = i + half*8, N = nbase + (l&15)
        // transposed store: tT[b][N][mloc + half*8 + i] -> 8 contiguous bf16
        int n = nbase + (l & 15);
        __bf16* dst = tT + ((size_t)b * kC + n) * kHW + mloc + half * 8;
        *(v8bf*)dst = v8f2bf(acc);
    }
}

// ---------------------------------------------------------------------------
// Kernel 4: fused  y = relu(xn xn^T)^2 @ t   (flash-attention style).
// Grid: (hw/64, B).  Block: 128 threads = 4 waves; wave w owns query rows
// [qblock*64 + w*16, +16) and ALL 256 output channels (16 f32 acc tiles).
// Per 32-key step: 16 WMMA (scores) + 16 WMMA (S*V); the V block (16 KB,
// identical for all 4 waves) is async-copied into double-buffered LDS.
// ---------------------------------------------------------------------------
__global__ void __launch_bounds__(128) ppm_attn_kernel(
        const __bf16* __restrict__ xn, const __bf16* __restrict__ tT,
        float* __restrict__ out) {
    __shared__ __bf16 sbuf[4][16][40];        // per-wave 16x32 S tile (+pad)
    __shared__ __bf16 vbuf[2][kC][40];        // double-buffered V block (+pad)
    int w = threadIdx.x >> 5;
    int l = threadIdx.x & 31;
    int half = l >> 4;
    int r = l & 15;
    int col = l & 15;
    int b = blockIdx.y;
    int qbase = blockIdx.x * 64 + w * 16;     // within batch
    const __bf16* xb = xn + (size_t)b * kHW * kC;   // [hw][C] row-major
    const __bf16* tb = tT + (size_t)b * kC * kHW;   // [C][hw] (transposed)

    // Query A-fragments: resident in registers for the whole kernel (64 VGPR)
    v16bf qfrag[8];
    #pragma unroll
    for (int kc = 0; kc < 8; ++kc) qfrag[kc] = load_a_g(xb, kC, qbase, kc * 32);

    v8f zero = {};
    v8f acc[16];
    #pragma unroll
    for (int nt = 0; nt < 16; ++nt) acc[nt] = zero;

    // cooperative async copy of a V block (256 ch x 32 px bf16 = 16 KB):
    // 1024 b128 chunks, 8 per thread, consecutive threads -> consecutive
    // chunks (coalesced); LDS rows padded to 40 elems for bank spread.
    auto issue_vcopy = [&](int jn, int nb) {
        #pragma unroll
        for (int i = 0; i < 8; ++i) {
            int q = (int)threadIdx.x + i * 128;
            int ch = q >> 2, part = q & 3;
            async_copy_b128(tb + (size_t)ch * kHW + jn + part * 8,
                            &vbuf[nb][ch][part * 8]);
        }
    };

    constexpr int nblk = kHW / 32;            // 98
    issue_vcopy(0, 0);
    for (int jb = 0; jb < nblk; ++jb) {
        int j = jb * 32;
        int buf = jb & 1;
        if (jb + 1 < nblk) {
            issue_vcopy(j + 32, buf ^ 1);     // prefetch next block into buf^1
            ppm_wait_async<8>();              // oldest 8 (current buf) done
        } else {
            ppm_wait_async<0>();
        }
        __syncthreads();                      // all waves' portions visible

        // prefetch next key block (32 rows x 512B) into WGP$: one row/lane
        if (jb + 1 < nblk) {
            const __bf16* nk = xb + (size_t)(j + 32 + l) * kC;
            __builtin_prefetch(nk, 0, 1);           // global_prefetch_b8
            __builtin_prefetch(nk + 128, 0, 1);     // +256B (rest of row)
        }

        // ---- scores: S = relu(Qn * Kn^T)^2, two 16x16 tiles (32 keys) ----
        #pragma unroll
        for (int st = 0; st < 2; ++st) {
            v8f s = zero;
            #pragma unroll
            for (int kc = 0; kc < 8; ++kc) {
                v16bf kf = load_b_g(xb, kC, j + st * 16, kc * 32);
                s = __builtin_amdgcn_wmma_f32_16x16x32_bf16(
                        false, qfrag[kc], false, kf, (short)0, s, false, false);
            }
            #pragma unroll
            for (int i = 0; i < 8; ++i) {
                float v = fmaxf(s[i], 0.f);
                s[i] = v * v;                  // relu(sim)^GAMMA, GAMMA=2
            }
            v8bf sb = v8f2bf(s);
            #pragma unroll
            for (int i = 0; i < 8; ++i)
                sbuf[w][i + half * 8][st * 16 + col] = sb[i];
        }
        // ---- reload S as a 16x32 A-fragment from LDS ----
        v16bf sfrag;
        {
            const __bf16* p = &sbuf[w][r][half * 8];
            v8bf lo = *(const v8bf*)p;          // K 0..7   / 8..15
            v8bf hi = *(const v8bf*)(p + 16);   // K 16..23 / 24..31
            sfrag = __builtin_shufflevector(lo, hi, 0,1,2,3,4,5,6,7,
                                                   8,9,10,11,12,13,14,15);
        }
        // ---- O += S * V : 16 output-channel tiles, V from LDS ----
        #pragma unroll
        for (int nt = 0; nt < 16; ++nt) {
            const __bf16* p = &vbuf[buf][nt * 16 + col][half * 8];
            v8bf lo = *(const v8bf*)p;
            v8bf hi = *(const v8bf*)(p + 16);
            v16bf vf = __builtin_shufflevector(lo, hi, 0,1,2,3,4,5,6,7,
                                                      8,9,10,11,12,13,14,15);
            acc[nt] = __builtin_amdgcn_wmma_f32_16x16x32_bf16(
                          false, sfrag, false, vf, (short)0, acc[nt], false, false);
        }
        __syncthreads();   // all waves done reading buf before it is refilled
    }

    // write y[b][m][c] (f32): per VGPR i, 16 lanes store contiguous channels
    #pragma unroll
    for (int nt = 0; nt < 16; ++nt) {
        int m = qbase + half * 8;
        float* dst = out + ((size_t)b * kHW + m) * kC + nt * 16 + col;
        #pragma unroll
        for (int i = 0; i < 8; ++i) dst[(size_t)i * kC] = acc[nt][i];
    }
}

// ---------------------------------------------------------------------------
extern "C" void kernel_launch(void* const* d_in, const int* in_sizes, int n_in,
                              void* d_out, int out_size, void* d_ws, size_t ws_size,
                              hipStream_t stream) {
    (void)in_sizes; (void)n_in; (void)out_size; (void)ws_size;
    const float* x = (const float*)d_in[0];
    const float* W = (const float*)d_in[1];
    float* out = (float*)d_out;

    char* ws = (char*)d_ws;
    size_t xn_bytes = (size_t)kRows * kC * 2;          // 12.8 MB
    size_t tT_bytes = (size_t)kB * kC * kHW * 2;       // 12.8 MB
    __bf16* xn = (__bf16*)ws;
    __bf16* tT = (__bf16*)(ws + xn_bytes);
    __bf16* Wt = (__bf16*)(ws + xn_bytes + tT_bytes);  // 128 KB

    ppm_norm_kernel <<<kHW, 256, 0, stream>>>(x, xn);
    ppm_wt_kernel   <<<kC, kC, 0, stream>>>(W, Wt);
    ppm_gemmt_kernel<<<kRows / 16, 128, 0, stream>>>(x, Wt, tT);
    dim3 grid(kHW / 64, kB);
    ppm_attn_kernel <<<grid, 128, 0, stream>>>(xn, tT, out);
}
